// LRPModel_85667417686427
// MI455X (gfx1250) — matrix operand order repaired and checked by reading, not compiled, not run on hardware
//
#include <hip/hip_runtime.h>
#include <hip/hip_bf16.h>
#include <math.h>

// ---- model dims ----
#define E_     128
#define H_     8
#define DK_    16
#define FFH_   512
#define NL_    6
#define DEPOT_ 20
#define CUST_  500
#define NN_    520
#define BB_    32
#define MM_    520
#define LL_    40
#define NP_    544   // keys padded to multiple of 32 for V^T

typedef __bf16 bf16_t;
typedef __attribute__((ext_vector_type(16))) __bf16 v16bf;
typedef __attribute__((ext_vector_type(8)))  float  v8f;

typedef union { v16bf v; uint4 u[2]; } fragU;

__device__ __forceinline__ v8f wmma_bf16(v16bf a, v16bf b, v8f c) {
  return __builtin_amdgcn_wmma_f32_16x16x32_bf16(false, a, false, b,
                                                 (short)0, c, false, false);
}

__device__ __forceinline__ uint4 uz4() { return make_uint4(0u, 0u, 0u, 0u); }

// branchless zero-mask (cndmask, no exec branches / no mid-branch waits)
__device__ __forceinline__ uint4 maskz(uint4 d, bool v) {
  d.x = v ? d.x : 0u;
  d.y = v ? d.y : 0u;
  d.z = v ? d.z : 0u;
  d.w = v ? d.w : 0u;
  return d;
}

// ---------------------------------------------------------------------------
// fp32 -> bf16 convert with optional K padding (rows x ksrc -> rows x kdst)
// ---------------------------------------------------------------------------
__global__ void k_convert_pad(const float* __restrict__ src, bf16_t* __restrict__ dst,
                              int rows, int ksrc, int kdst) {
  int idx = blockIdx.x * blockDim.x + threadIdx.x;
  int total = rows * kdst;
  if (idx >= total) return;
  int r = idx / kdst, c = idx - r * kdst;
  float v = (c < ksrc) ? src[(size_t)r * ksrc + c] : 0.f;
  dst[idx] = (bf16_t)v;
}

// ---------------------------------------------------------------------------
// embedding: depot (B,20,4)@Wd^T + bd ; customer (B,500,3)@Wc^T + bc
// ---------------------------------------------------------------------------
__global__ void k_embed(const float* __restrict__ dep, const float* __restrict__ cus,
                        const float* __restrict__ Wd, const float* __restrict__ bd,
                        const float* __restrict__ Wc, const float* __restrict__ bc,
                        float* __restrict__ xf, bf16_t* __restrict__ xb) {
  int idx = blockIdx.x * blockDim.x + threadIdx.x;
  int total = BB_ * NN_ * E_;
  if (idx >= total) return;
  int e = idx % E_;
  int bn = idx / E_;
  int n = bn % NN_, b = bn / NN_;
  float v;
  if (n < DEPOT_) {
    const float* f = dep + ((size_t)b * DEPOT_ + n) * 4;
    v = bd[e];
    #pragma unroll
    for (int j = 0; j < 4; ++j) v += f[j] * Wd[e * 4 + j];
  } else {
    const float* f = cus + ((size_t)b * CUST_ + (n - DEPOT_)) * 3;
    v = bc[e];
    #pragma unroll
    for (int j = 0; j < 3; ++j) v += f[j] * Wc[e * 3 + j];
  }
  xf[idx] = v;
  xb[idx] = (bf16_t)v;
}

// ---------------------------------------------------------------------------
// per-head V transpose with zero pad: Vt[b,h,d,n] (n padded to NP_)
// ---------------------------------------------------------------------------
__global__ void k_vt(const bf16_t* __restrict__ V, bf16_t* __restrict__ Vt) {
  int idx = blockIdx.x * blockDim.x + threadIdx.x;
  int total = BB_ * H_ * DK_ * NP_;
  if (idx >= total) return;
  int n = idx % NP_;
  int t = idx / NP_;
  int d = t % DK_;
  int bh = t / DK_;
  int h = bh % H_, b = bh / H_;
  bf16_t v = (bf16_t)0.f;
  if (n < NN_) v = V[((size_t)b * NN_ + n) * E_ + h * DK_ + d];
  Vt[idx] = v;
}

// ---------------------------------------------------------------------------
// Batched WMMA GEMM:  Y[b] = A[b] (R x K) @ W[b]^T  (W: Nout x K rowmajor)
// wave -> 32x64 strip (2 m-tiles x 4 n-tiles): 8 wmma per 12 b128 loads.
// GUARD=false fast path has no edge checks (R%32==0 && Nout%64==0).
// ---------------------------------------------------------------------------
template <bool GUARD>
__global__ void k_gemm_bt(const bf16_t* __restrict__ A, long long sA,
                          const bf16_t* __restrict__ W, long long sW,
                          const float* __restrict__ bias,
                          float* __restrict__ Yf, bf16_t* __restrict__ Yb, long long sY,
                          int R, int K, int Nout, int relu) {
  const bf16_t* Ab = A + (size_t)blockIdx.y * (size_t)sA;
  const bf16_t* Wb = W + (size_t)blockIdx.y * (size_t)sW;
  int lane = threadIdx.x & 31;
  int wave = blockIdx.x * (blockDim.x >> 5) + (threadIdx.x >> 5);
  int tiles_m = (R + 31) >> 5;
  int tiles_n = (Nout + 63) >> 6;
  if (wave >= tiles_m * tiles_n) return;
  int mt = wave / tiles_n, nt = wave % tiles_n;
  int m0 = mt << 5;
  int half = lane >> 4, lh = lane & 15;

  bool avalid[2];
  const bf16_t* arow[2];
  #pragma unroll
  for (int i = 0; i < 2; ++i) {
    int rowi = m0 + i * 16 + lh;
    avalid[i] = !GUARD || (rowi < R);
    arow[i] = Ab + (size_t)(avalid[i] ? rowi : 0) * K;
  }
  bool bvalid[4];
  const bf16_t* wrow[4];
  #pragma unroll
  for (int t = 0; t < 4; ++t) {
    int coln = (nt * 4 + t) * 16 + lh;
    bvalid[t] = !GUARD || (coln < Nout);
    wrow[t] = Wb + (size_t)(bvalid[t] ? coln : 0) * K;
  }
  v8f acc[2][4];
  #pragma unroll
  for (int i = 0; i < 2; ++i)
    #pragma unroll
    for (int t = 0; t < 4; ++t) acc[i][t] = (v8f){};

  for (int k0 = 0; k0 < K; k0 += 32) {
    fragU a[2];
    #pragma unroll
    for (int i = 0; i < 2; ++i) {
      const uint4* ap_ = (const uint4*)(arow[i] + k0 + 8 * half);
      uint4 a0 = ap_[0], a1 = ap_[2];  // segs at +0 and +16 elements
      if (GUARD) { a0 = maskz(a0, avalid[i]); a1 = maskz(a1, avalid[i]); }
      a[i].u[0] = a0; a[i].u[1] = a1;
    }
    #pragma unroll
    for (int t = 0; t < 4; ++t) {
      fragU bm;
      const uint4* bp_ = (const uint4*)(wrow[t] + k0 + 8 * half);
      uint4 b0 = bp_[0], b1 = bp_[2];  // segs at +0 and +16 elements
      if (GUARD) { b0 = maskz(b0, bvalid[t]); b1 = maskz(b1, bvalid[t]); }
      bm.u[0] = b0; bm.u[1] = b1;
      acc[0][t] = wmma_bf16(a[0].v, bm.v, acc[0][t]);
      acc[1][t] = wmma_bf16(a[1].v, bm.v, acc[1][t]);
    }
  }

  #pragma unroll
  for (int t = 0; t < 4; ++t) {
    int col = (nt * 4 + t) * 16 + lh;
    if (GUARD && col >= Nout) continue;
    float bv = bias ? bias[col] : 0.f;
    #pragma unroll
    for (int i = 0; i < 2; ++i) {
      #pragma unroll
      for (int r = 0; r < 8; ++r) {  // C: row = r + 8*half, col = lh
        int row = m0 + i * 16 + r + 8 * half;
        if (GUARD && row >= R) continue;
        float v = acc[i][t][r] + bv;
        if (relu) v = fmaxf(v, 0.f);
        size_t off = (size_t)blockIdx.y * (size_t)sY + (size_t)row * Nout + col;
        if (Yf) Yf[off] = v;
        if (Yb) Yb[off] = (bf16_t)v;
      }
    }
  }
}

// ---------------------------------------------------------------------------
// Flash attention, 32 keys/iteration. One wave per (b,h,16-query-row tile).
// Heads packed in E. scale = 1/sqrt(dk). Vt = per-head transposed V (B,H,DK,NP_).
// HAS_MASK: additive mask (B,Mq,Nk) shared by all heads, loaded branchlessly.
// ---------------------------------------------------------------------------
template <bool HAS_MASK>
__global__ void k_flash(const bf16_t* __restrict__ Qb, const bf16_t* __restrict__ Kb,
                        const bf16_t* __restrict__ Vt, const float* __restrict__ maskf,
                        bf16_t* __restrict__ Ob, int Mq, int Nk) {
  __shared__ __align__(16) bf16_t psh[16 * 32];
  int lane = threadIdx.x & 31;
  int half = lane >> 4, lh = lane & 15;
  int m0 = blockIdx.x << 4;
  int b = blockIdx.y / H_, h = blockIdx.y % H_;
  const float scale = 0.25f;  // 1/sqrt(16)

  bool aval = (m0 + lh) < Mq;
  fragU aq;
  {
    const bf16_t* qrow = Qb + ((size_t)b * Mq + (aval ? (m0 + lh) : 0)) * E_ + h * DK_;
    const uint4* qp = (const uint4*)(qrow + 8 * half);  // dk 8*half..+7, always < 16
    aq.u[0] = maskz(qp[0], aval);
    aq.u[1] = uz4();  // dk pad 16..31
  }
  const bf16_t* vtrow = Vt + ((size_t)(b * H_ + h) * DK_ + lh) * NP_;
  const float* mbase = HAS_MASK ? (maskf + (size_t)b * Mq * Nk) : nullptr;

  float mrow[8], lrow[8];
  v8f oacc = (v8f){};
  #pragma unroll
  for (int r = 0; r < 8; ++r) { mrow[r] = -1e30f; lrow[r] = 0.f; }

  int ktiles = (Nk + 31) >> 5;
  for (int kt = 0; kt < ktiles; ++kt) {
    int key0 = kt << 5;

    // branchless mask prefetch: 16 unconditional b32 loads through clamped
    // addresses -> compiler can clause them and overlap with the score WMMAs
    float mval[2][8];
    if (HAS_MASK) {
      #pragma unroll
      for (int sub = 0; sub < 2; ++sub) {
        int key = key0 + sub * 16 + lh;
        int kc = (key < Nk) ? key : 0;
        #pragma unroll
        for (int r = 0; r < 8; ++r) {
          int row = m0 + r + 8 * half;
          int rc = (row < Mq) ? row : 0;
          float mv = mbase[(size_t)rc * Nk + kc];
          mval[sub][r] = ((key < Nk) && (row < Mq)) ? mv : 0.f;
        }
      }
    }

    float sv[2][8];
    #pragma unroll
    for (int sub = 0; sub < 2; ++sub) {
      int key = key0 + sub * 16 + lh;
      bool kv = (key < Nk);  // dk 8*half..+7 in u[0], dk pad in u[1]
      const uint4* kp =
          (const uint4*)(Kb + ((size_t)b * Nk + ((key < Nk) ? key : 0)) * E_ + h * DK_ + 8 * half);
      fragU bk;
      bk.u[0] = maskz(kp[0], kv);
      bk.u[1] = uz4();
      v8f s = wmma_bf16(aq.v, bk.v, (v8f){});
      #pragma unroll
      for (int r = 0; r < 8; ++r) {
        float v = s[r] * scale;
        if (HAS_MASK) v += mval[sub][r];
        if (key >= Nk) v = -1e30f;
        sv[sub][r] = v;
      }
    }
    // online softmax per row; 16-lane reductions stay within each half
    #pragma unroll
    for (int r = 0; r < 8; ++r) {
      float mx = fmaxf(sv[0][r], sv[1][r]);
      mx = fmaxf(mx, __shfl_xor(mx, 1, 32));
      mx = fmaxf(mx, __shfl_xor(mx, 2, 32));
      mx = fmaxf(mx, __shfl_xor(mx, 4, 32));
      mx = fmaxf(mx, __shfl_xor(mx, 8, 32));
      float mnew = fmaxf(mrow[r], mx);
      float p0 = __expf(sv[0][r] - mnew);
      float p1 = __expf(sv[1][r] - mnew);
      float ps = p0 + p1;
      ps += __shfl_xor(ps, 1, 32);
      ps += __shfl_xor(ps, 2, 32);
      ps += __shfl_xor(ps, 4, 32);
      ps += __shfl_xor(ps, 8, 32);
      float a_ = __expf(mrow[r] - mnew);
      lrow[r] = lrow[r] * a_ + ps;
      mrow[r] = mnew;
      oacc[r] *= a_;
      sv[0][r] = p0;
      sv[1][r] = p1;
    }
    // stage P (16x32) in LDS to re-enter A-fragment layout
    #pragma unroll
    for (int r = 0; r < 8; ++r) {
      psh[(r + 8 * half) * 32 + lh] = (bf16_t)sv[0][r];
      psh[(r + 8 * half) * 32 + 16 + lh] = (bf16_t)sv[1][r];
    }
    __syncthreads();
    fragU ap;
    {
      const uint4* pp = (const uint4*)(psh + lh * 32 + 8 * half);
      ap.u[0] = pp[0];   // keys 8*half..+7
      ap.u[1] = pp[2];   // keys 16+8*half..+7
    }
    __syncthreads();
    fragU bv;
    {
      const uint4* vp = (const uint4*)(vtrow + key0 + 8 * half);  // zero-padded
      bv.u[0] = vp[0];
      bv.u[1] = vp[2];
    }
    oacc = wmma_bf16(ap.v, bv.v, oacc);  // full K=32, no wasted MACs
  }

  #pragma unroll
  for (int r = 0; r < 8; ++r) {
    int row = m0 + r + 8 * half;
    if (row < Mq) {
      float v = oacc[r] / fmaxf(lrow[r], 1e-20f);
      Ob[((size_t)b * Mq + row) * E_ + h * DK_ + lh] = (bf16_t)v;
    }
  }
}

// ---------------------------------------------------------------------------
// InstanceNorm over node axis with fused residual: y = inorm(A + D)*g + b
// ---------------------------------------------------------------------------
__global__ void k_inorm(const float* __restrict__ A, const float* __restrict__ D,
                        const float* __restrict__ g, const float* __restrict__ bta,
                        float* __restrict__ Yf, bf16_t* __restrict__ Yb) {
  int lane = threadIdx.x & 31;
  int wave = blockIdx.x * (blockDim.x >> 5) + (threadIdx.x >> 5);
  if (wave >= BB_ * E_) return;
  int b = wave / E_, e = wave % E_;
  float v[17];
  float s = 0.f, s2 = 0.f;
  #pragma unroll
  for (int i = 0; i < 17; ++i) {
    int n = i * 32 + lane;
    float x = 0.f;
    if (n < NN_) {
      size_t off = ((size_t)b * NN_ + n) * E_ + e;
      x = A[off] + D[off];
    }
    v[i] = x; s += x; s2 += x * x;
  }
  #pragma unroll
  for (int o = 1; o < 32; o <<= 1) {
    s += __shfl_xor(s, o, 32);
    s2 += __shfl_xor(s2, o, 32);
  }
  float mean = s / (float)NN_;
  float var = s2 / (float)NN_ - mean * mean;
  float inv = rsqrtf(var + 1e-5f) * g[e];
  float bb = bta[e];
  #pragma unroll
  for (int i = 0; i < 17; ++i) {
    int n = i * 32 + lane;
    if (n < NN_) {
      float y = (v[i] - mean) * inv + bb;
      size_t off = ((size_t)b * NN_ + n) * E_ + e;
      Yf[off] = y;
      Yb[off] = (bf16_t)y;
    }
  }
}

// mean over nodes: out(B,E) = mean_n X(B,N,E)
__global__ void k_rowmean(const float* __restrict__ X, float* __restrict__ out) {
  int lane = threadIdx.x & 31;
  int wave = blockIdx.x * (blockDim.x >> 5) + (threadIdx.x >> 5);
  if (wave >= BB_ * E_) return;
  int b = wave / E_, e = wave % E_;
  float s = 0.f;
  #pragma unroll
  for (int i = 0; i < 17; ++i) {
    int n = i * 32 + lane;
    if (n < NN_) s += X[((size_t)b * NN_ + n) * E_ + e];
  }
  #pragma unroll
  for (int o = 1; o < 32; o <<= 1) s += __shfl_xor(s, o, 32);
  if (lane == 0) out[(size_t)b * E_ + e] = s / (float)NN_;
}

// ---------------------------------------------------------------------------
// build decoder query inputs: one wave per (b,m)
// ---------------------------------------------------------------------------
__global__ void k_build_q(const float* __restrict__ xf, const float* __restrict__ encmean,
                          const float* __restrict__ loadv, const int* __restrict__ cur,
                          const int* __restrict__ subn, const int* __restrict__ subl,
                          bf16_t* __restrict__ qrout, bf16_t* __restrict__ qloc) {
  int lane = threadIdx.x & 31;
  int wave = blockIdx.x * (blockDim.x >> 5) + (threadIdx.x >> 5);
  if (wave >= BB_ * MM_) return;
  int b = wave / MM_, m = wave % MM_;
  int cn = cur[(size_t)b * MM_ + m];
  const float* lrow = xf + ((size_t)b * NN_ + cn) * E_;
  float le[4], em[4], sm[4] = {0.f, 0.f, 0.f, 0.f};
  #pragma unroll
  for (int j = 0; j < 4; ++j) {
    int e = lane + 32 * j;
    le[j] = lrow[e];
    em[j] = encmean[(size_t)b * E_ + e];
  }
  int slen = subl[(size_t)b * MM_ + m];
  const int* srow = subn + ((size_t)b * MM_ + m) * LL_;
  float cnt = 0.f;
  for (int l = 0; l < LL_; ++l) {
    int nd = srow[l];
    if (l < slen && nd >= DEPOT_) {
      cnt += 1.f;
      const float* er = xf + ((size_t)b * NN_ + nd) * E_;
      #pragma unroll
      for (int j = 0; j < 4; ++j) sm[j] += er[lane + 32 * j];
    }
  }
  float icnt = 1.f / fmaxf(cnt, 1.f);
  size_t r = (size_t)b * MM_ + m;
  bf16_t* qr = qrout + r * 288;
  bf16_t* ql = qloc + r * 256;
  #pragma unroll
  for (int j = 0; j < 4; ++j) {
    int e = lane + 32 * j;
    qr[e] = (bf16_t)le[j];
    qr[128 + e] = (bf16_t)em[j];
    ql[e] = (bf16_t)le[j];
    ql[128 + e] = (bf16_t)(sm[j] * icnt);
  }
  if (lane == 0) qr[256] = (bf16_t)loadv[(size_t)b * MM_ + m];
  int p = 257 + lane;
  if (p < 288) qr[p] = (bf16_t)0.f;
}

// final_q = flag ? q_location : q_routing  (bf16 out)
__global__ void k_blend(const float* __restrict__ qloc, const float* __restrict__ qrout,
                        const int* __restrict__ cur, const int* __restrict__ sel,
                        bf16_t* __restrict__ fq) {
  int idx = blockIdx.x * blockDim.x + threadIdx.x;
  int total = BB_ * MM_ * E_;
  if (idx >= total) return;
  int bm = idx / E_;
  bool last_depot = sel[(size_t)bm * 4 + 2] < DEPOT_;
  bool node_depot = cur[bm] < DEPOT_;
  bool flag = (!last_depot) && node_depot;
  fq[idx] = (bf16_t)(flag ? qloc[idx] : qrout[idx]);
}

// final: prob = softmax_n( 10*tanh(score/sqrt(E)) + mask ), in place on d_out
__global__ void k_final_softmax(float* __restrict__ out, const float* __restrict__ maskf) {
  int lane = threadIdx.x & 31;
  int wave = blockIdx.x * (blockDim.x >> 5) + (threadIdx.x >> 5);
  if (wave >= BB_ * MM_) return;
  float* row = out + (size_t)wave * NN_;
  const float* mrow = maskf + (size_t)wave * NN_;
  const float inv_sqrtE = 0.08838834764831845f;
  float v[17];
  float mx = -1e30f;
  #pragma unroll
  for (int i = 0; i < 17; ++i) {
    int n = i * 32 + lane;
    float x = -1e30f;
    if (n < NN_) x = 10.f * tanhf(row[n] * inv_sqrtE) + mrow[n];
    v[i] = x;
    mx = fmaxf(mx, x);
  }
  #pragma unroll
  for (int o = 1; o < 32; o <<= 1) mx = fmaxf(mx, __shfl_xor(mx, o, 32));
  float s = 0.f;
  #pragma unroll
  for (int i = 0; i < 17; ++i) {
    int n = i * 32 + lane;
    if (n < NN_) { v[i] = __expf(v[i] - mx); s += v[i]; }
  }
  #pragma unroll
  for (int o = 1; o < 32; o <<= 1) s += __shfl_xor(s, o, 32);
  float is = 1.f / s;
  #pragma unroll
  for (int i = 0; i < 17; ++i) {
    int n = i * 32 + lane;
    if (n < NN_) row[n] = v[i] * is;
  }
}

// ---------------------------------------------------------------------------
// host side
// ---------------------------------------------------------------------------
static void launch_gemm(const bf16_t* A, long long sA, const bf16_t* W, long long sW,
                        const float* bias, float* Yf, bf16_t* Yb, long long sY,
                        int R, int K, int Nout, int relu, int batch, hipStream_t s) {
  int tiles_m = (R + 31) >> 5;
  int tiles_n = (Nout + 63) >> 6;
  int tiles = tiles_m * tiles_n;
  dim3 grid((tiles + 3) / 4, batch);
  bool guard = (R & 31) || (Nout & 63);
  if (guard)
    k_gemm_bt<true><<<grid, 128, 0, s>>>(A, sA, W, sW, bias, Yf, Yb, sY, R, K, Nout, relu);
  else
    k_gemm_bt<false><<<grid, 128, 0, s>>>(A, sA, W, sW, bias, Yf, Yb, sY, R, K, Nout, relu);
}

static void launch_convert(const float* src, bf16_t* dst, int rows, int ksrc, int kdst,
                           hipStream_t s) {
  int total = rows * kdst;
  k_convert_pad<<<(total + 255) / 256, 256, 0, s>>>(src, dst, rows, ksrc, kdst);
}

extern "C" void kernel_launch(void* const* d_in, const int* in_sizes, int n_in,
                              void* d_out, int out_size, void* d_ws, size_t ws_size,
                              hipStream_t stream) {
  (void)in_sizes; (void)n_in; (void)out_size; (void)ws_size;
  const float* depot = (const float*)d_in[0];
  const float* cust  = (const float*)d_in[1];
  const float* maskf = (const float*)d_in[2];
  const float* loadv = (const float*)d_in[3];
  const int*   cur   = (const int*)d_in[4];
  const int*   subn  = (const int*)d_in[5];
  const int*   subl  = (const int*)d_in[6];
  const int*   sel   = (const int*)d_in[7];
  const float* Wd    = (const float*)d_in[8];
  const float* bd    = (const float*)d_in[9];
  const float* Wcu   = (const float*)d_in[10];
  const float* bcu   = (const float*)d_in[11];
  const float* Wq    = (const float*)d_in[12];
  const float* Wk    = (const float*)d_in[13];
  const float* Wv    = (const float*)d_in[14];
  const float* Wc    = (const float*)d_in[15];
  const float* Wc_b  = (const float*)d_in[16];
  const float* n1g   = (const float*)d_in[17];
  const float* n1b   = (const float*)d_in[18];
  const float* fW1   = (const float*)d_in[19];
  const float* fb1   = (const float*)d_in[20];
  const float* fW2   = (const float*)d_in[21];
  const float* fb2   = (const float*)d_in[22];
  const float* n2g   = (const float*)d_in[23];
  const float* n2b   = (const float*)d_in[24];
  const float* dWql  = (const float*)d_in[25];
  const float* dWqr  = (const float*)d_in[26];
  const float* dWk   = (const float*)d_in[27];
  const float* dWv   = (const float*)d_in[28];
  const float* dWc   = (const float*)d_in[29];
  const float* dWcb  = (const float*)d_in[30];
  float* prob = (float*)d_out;

  const size_t R = (size_t)BB_ * NN_;  // 16640 rows

  char* ws = (char*)d_ws;
  size_t off = 0;
  auto alloc = [&](size_t bytes) -> void* {
    void* p = ws + off;
    off += (bytes + 255) & ~(size_t)255;
    return p;
  };
  float*  xf   = (float*)alloc(R * E_ * 4);
  float*  x1f  = (float*)alloc(R * E_ * 4);
  float*  mhf  = (float*)alloc(R * E_ * 4);  // also q_routing fp32
  float*  fff  = (float*)alloc(R * E_ * 4);  // also q_location fp32
  bf16_t* xb   = (bf16_t*)alloc(R * E_ * 2);
  bf16_t* x1b  = (bf16_t*)alloc(R * E_ * 2);  // also score bf16
  bf16_t* qb   = (bf16_t*)alloc(R * E_ * 2);  // also final_q
  bf16_t* kb   = (bf16_t*)alloc(R * E_ * 2);  // also k_d
  bf16_t* vb   = (bf16_t*)alloc(R * E_ * 2);  // also v_d
  bf16_t* ob   = (bf16_t*)alloc(R * E_ * 2);  // mha / dec-attn output
  bf16_t* vtb  = (bf16_t*)alloc((size_t)BB_ * H_ * DK_ * NP_ * 2);  // V^T per head
  bf16_t* h1b  = (bf16_t*)alloc(R * FFH_ * 2);
  float*  encm = (float*)alloc((size_t)BB_ * E_ * 4);
  bf16_t* qcr  = (bf16_t*)alloc(R * 288 * 2);
  bf16_t* qcl  = (bf16_t*)alloc(R * 256 * 2);
  bf16_t* wqb  = (bf16_t*)alloc((size_t)NL_ * E_ * E_ * 2);
  bf16_t* wkb  = (bf16_t*)alloc((size_t)NL_ * E_ * E_ * 2);
  bf16_t* wvb  = (bf16_t*)alloc((size_t)NL_ * E_ * E_ * 2);
  bf16_t* wcb  = (bf16_t*)alloc((size_t)NL_ * E_ * E_ * 2);
  bf16_t* w1b  = (bf16_t*)alloc((size_t)NL_ * FFH_ * E_ * 2);
  bf16_t* w2b  = (bf16_t*)alloc((size_t)NL_ * E_ * FFH_ * 2);
  bf16_t* dql  = (bf16_t*)alloc((size_t)E_ * 256 * 2);
  bf16_t* dqr  = (bf16_t*)alloc((size_t)E_ * 288 * 2);  // K padded 257->288
  bf16_t* dkb  = (bf16_t*)alloc((size_t)E_ * E_ * 2);
  bf16_t* dvb  = (bf16_t*)alloc((size_t)E_ * E_ * 2);
  bf16_t* dcb  = (bf16_t*)alloc((size_t)E_ * E_ * 2);

  // ---- weight conversion ----
  launch_convert(Wq,  wqb, NL_ * E_,   E_,   E_,   stream);
  launch_convert(Wk,  wkb, NL_ * E_,   E_,   E_,   stream);
  launch_convert(Wv,  wvb, NL_ * E_,   E_,   E_,   stream);
  launch_convert(Wc,  wcb, NL_ * E_,   E_,   E_,   stream);
  launch_convert(fW1, w1b, NL_ * FFH_, E_,   E_,   stream);
  launch_convert(fW2, w2b, NL_ * E_,   FFH_, FFH_, stream);
  launch_convert(dWql, dql, E_, 256, 256, stream);
  launch_convert(dWqr, dqr, E_, 257, 288, stream);
  launch_convert(dWk,  dkb, E_, E_, E_, stream);
  launch_convert(dWv,  dvb, E_, E_, E_, stream);
  launch_convert(dWc,  dcb, E_, E_, E_, stream);

  // ---- embedding ----
  {
    int total = BB_ * NN_ * E_;
    k_embed<<<(total + 255) / 256, 256, 0, stream>>>(depot, cust, Wd, bd, Wcu, bcu, xf, xb);
  }

  dim3 flash_grid((MM_ + 15) / 16, BB_ * H_);
  int wgrid = (BB_ * E_ + 3) / 4;    // 4 waves / block
  int bmgrid = (BB_ * MM_ + 3) / 4;
  int vt_total = BB_ * H_ * DK_ * NP_;
  int vt_grid = (vt_total + 255) / 256;

  // ---- encoder layers ----
  for (int i = 0; i < NL_; ++i) {
    size_t wo = (size_t)i * E_ * E_;
    launch_gemm(xb, 0, wqb + wo, 0, nullptr, nullptr, qb, 0, (int)R, E_, E_, 0, 1, stream);
    launch_gemm(xb, 0, wkb + wo, 0, nullptr, nullptr, kb, 0, (int)R, E_, E_, 0, 1, stream);
    launch_gemm(xb, 0, wvb + wo, 0, nullptr, nullptr, vb, 0, (int)R, E_, E_, 0, 1, stream);
    k_vt<<<vt_grid, 256, 0, stream>>>(vb, vtb);
    k_flash<false><<<flash_grid, 32, 0, stream>>>(qb, kb, vtb, nullptr, ob, NN_, NN_);
    launch_gemm(ob, 0, wcb + wo, 0, Wc_b + (size_t)i * E_, mhf, nullptr, 0,
                (int)R, E_, E_, 0, 1, stream);
    k_inorm<<<wgrid, 128, 0, stream>>>(xf, mhf, n1g + (size_t)i * E_, n1b + (size_t)i * E_,
                                       x1f, x1b);
    launch_gemm(x1b, 0, w1b + (size_t)i * FFH_ * E_, 0, fb1 + (size_t)i * FFH_,
                nullptr, h1b, 0, (int)R, E_, FFH_, 1, 1, stream);
    launch_gemm(h1b, 0, w2b + (size_t)i * E_ * FFH_, 0, fb2 + (size_t)i * E_,
                fff, nullptr, 0, (int)R, FFH_, E_, 0, 1, stream);
    k_inorm<<<wgrid, 128, 0, stream>>>(x1f, fff, n2g + (size_t)i * E_, n2b + (size_t)i * E_,
                                       xf, xb);
  }

  // ---- decoder k/v ----
  launch_gemm(xb, 0, dkb, 0, nullptr, nullptr, kb, 0, (int)R, E_, E_, 0, 1, stream);
  launch_gemm(xb, 0, dvb, 0, nullptr, nullptr, vb, 0, (int)R, E_, E_, 0, 1, stream);
  k_vt<<<vt_grid, 256, 0, stream>>>(vb, vtb);

  // ---- decoder queries ----
  k_rowmean<<<wgrid, 128, 0, stream>>>(xf, encm);
  k_build_q<<<bmgrid, 128, 0, stream>>>(xf, encm, loadv, cur, subn, subl, qcr, qcl);
  launch_gemm(qcr, 0, dqr, 0, nullptr, mhf, nullptr, 0, (int)R, 288, E_, 0, 1, stream); // q_routing
  launch_gemm(qcl, 0, dql, 0, nullptr, fff, nullptr, 0, (int)R, 256, E_, 0, 1, stream); // q_location
  {
    int total = BB_ * MM_ * E_;
    k_blend<<<(total + 255) / 256, 256, 0, stream>>>(fff, mhf, cur, sel, qb);
  }

  // ---- decoder attention (with additive mask) ----
  k_flash<true><<<flash_grid, 32, 0, stream>>>(qb, kb, vtb, maskf, ob, MM_, NN_);

  // ---- score projection, compatibility with nodes, final softmax ----
  launch_gemm(ob, 0, dcb, 0, dWcb, nullptr, x1b, 0, (int)R, E_, E_, 0, 1, stream);
  // score_nodes[b] = score[b] (M,E) @ encoded[b]^T -> directly into d_out (B,M,N)
  launch_gemm(x1b, (long long)MM_ * E_, xb, (long long)NN_ * E_, nullptr,
              prob, nullptr, (long long)MM_ * NN_, MM_, E_, NN_, 0, BB_, stream);
  k_final_softmax<<<bmgrid, 128, 0, stream>>>(prob, maskf);
}
